// GraphTransformer_2817498546751
// MI455X (gfx1250) — compile-verified, matrix operand order checked by reading, not run on hardware
//
#include <hip/hip_runtime.h>
#include <hip/hip_bf16.h>

typedef __attribute__((ext_vector_type(16))) _Float16 v16h;
typedef __attribute__((ext_vector_type(8)))  _Float16 v8h;
typedef __attribute__((ext_vector_type(4)))  _Float16 v4h;
typedef __attribute__((ext_vector_type(8)))  float    v8f;

// ---------------- WMMA fragment helpers (CDNA5 16x16x32 f16 layouts) ----------

// A fragment: 16x32 tile from row-major f16 LDS (leading dim ld).
// Per-lane data is two contiguous 8-element runs -> 2x ds_load_b128.
__device__ __forceinline__ v16h load_a_lds(const _Float16* As, int ld, int k0, int lane) {
    int r = lane & 15, h = lane >> 4;
    const _Float16* p = As + r * ld + k0 + h * 8;
    v8h lo = *(const v8h*)p;          // K = k0 + h*8 .. +7
    v8h hi = *(const v8h*)(p + 16);   // K = k0 + 16 + h*8 .. +7
    return __builtin_shufflevector(lo, hi, 0, 1, 2, 3, 4, 5, 6, 7,
                                   8, 9, 10, 11, 12, 13, 14, 15);
}

// B fragment from a pre-swizzled LDS panel: fragment-major, 16 contiguous f16
// per lane -> 2x ds_load_b128.
__device__ __forceinline__ v16h load_b_frag(const _Float16* Bs, int frag, int lane) {
    const v8h* p = (const v8h*)(Bs + (((frag << 5) + lane) << 4));
    v8h lo = p[0], hi = p[1];
    return __builtin_shufflevector(lo, hi, 0, 1, 2, 3, 4, 5, 6, 7,
                                   8, 9, 10, 11, 12, 13, 14, 15);
}

// D store: VGPR r holds row r + 8*(lane>>4), col = lane&15.
__device__ __forceinline__ void store_d(float* out, int ld, int row0, int n0, v8f c,
                                        int lane, const float* bias) {
    int n = lane & 15, h = lane >> 4;
    float bv = bias ? bias[n0 + n] : 0.0f;
#pragma unroll
    for (int r = 0; r < 8; ++r) {
        int row = row0 + r + 8 * h;
        out[(size_t)row * ld + n0 + n] = c[r] + bv;
    }
}

#define WMMA_F16(a, b, c) \
    __builtin_amdgcn_wmma_f32_16x16x32_f16(false, (a), false, (b), (short)0, (c), false, false)

// Stage activation rows (f32 row-major, 2^lcol cols) into f16 LDS, float4-vectorized.
__device__ __forceinline__ void stage_a(_Float16* As, const float* __restrict__ src,
                                        int rows, int lcol, int row0, int tid, int nt) {
    int total4 = (rows << lcol) >> 2;
    for (int i = tid; i < total4; i += nt) {
        int r  = i >> (lcol - 2);
        int c4 = i & ((1 << (lcol - 2)) - 1);
        float4 t = *((const float4*)(src + ((size_t)(row0 + r) << lcol)) + c4);
        v4h hv;
        hv[0] = (_Float16)t.x; hv[1] = (_Float16)t.y;
        hv[2] = (_Float16)t.z; hv[3] = (_Float16)t.w;
        ((v4h*)As)[i] = hv;
    }
}

// Stage weight panel W[K x 2^lcol] (f32 row-major) into LDS pre-swizzled to
// WMMA-B fragment-major layout: frag = kt*(Ncols/16)+nt, then [lane][16 f16].
// Global reads coalesced; LDS writes scattered (banked, cheap).
__device__ __forceinline__ void stage_b_sw(_Float16* Bs, const float* __restrict__ W,
                                           int K, int lcol, int tid, int nt) {
    int Ncols = 1 << lcol;
    int NT = Ncols >> 4;
    int total = K << lcol;
    for (int j = tid; j < total; j += nt) {
        int kg = j >> lcol;
        int n  = j & (Ncols - 1);
        int kt = kg >> 5, kk = kg & 31;
        int bit  = kk & 1;
        int hi   = (kk >= 16) ? 1 : 0;
        int rem  = kk - (hi << 4);
        int half = (rem >= 8) ? 1 : 0;
        int r2   = rem - (half << 3);
        int v    = (hi << 2) + (r2 >> 1);
        int e    = (v << 1) + bit;
        int lane = (half << 4) + (n & 15);
        int frag = kt * NT + (n >> 4);
        Bs[(((frag << 5) + lane) << 4) + e] = (_Float16)W[j];
    }
}

// ---------------- GEMM kernels (32 rows/block, 256 threads = 8 waves) ---------

// hx[N,64] = x[N,256] @ W1[256,64]
__global__ void gemm_x_w1(const float* __restrict__ x, const float* __restrict__ W1,
                          float* __restrict__ hx) {
    __shared__ __align__(32) _Float16 As[32 * 256];   // 16 KB
    __shared__ __align__(32) _Float16 Bs[256 * 64];   // 32 KB swizzled
    int tid = threadIdx.x;
    int row0 = blockIdx.x * 32;
    stage_a(As, x, 32, 8, row0, tid, 256);
    stage_b_sw(Bs, W1, 256, 6, tid, 256);
    __syncthreads();

    int wave = tid >> 5, lane = tid & 31;
    int rt = wave >> 2, ct = wave & 3;                // 2 row-tiles x 4 col-tiles
    const _Float16* Aw = As + rt * 16 * 256;
    v8f acc = {};
#pragma unroll
    for (int kt = 0; kt < 8; ++kt) {
        v16h a = load_a_lds(Aw, 256, kt * 32, lane);
        v16h b = load_b_frag(Bs, kt * 4 + ct, lane);
        acc = WMMA_F16(a, b, acc);
    }
    store_d(hx, 64, row0 + rt * 16, ct * 16, acc, lane, nullptr);
}

// q/k/v/hs[N,64] = h1[N,64] @ {Wq,Wk,Wv,Ws} + bias
__global__ void gemm_qkvs(const float* __restrict__ h1,
                          const float* __restrict__ Wq, const float* __restrict__ bq,
                          const float* __restrict__ Wk, const float* __restrict__ bk,
                          const float* __restrict__ Wv, const float* __restrict__ bv,
                          const float* __restrict__ Ws, const float* __restrict__ bs,
                          float* __restrict__ q, float* __restrict__ k,
                          float* __restrict__ v, float* __restrict__ hs) {
    __shared__ __align__(32) _Float16 As[32 * 64];       // 4 KB
    __shared__ __align__(32) _Float16 Bs[4 * 64 * 64];   // 32 KB swizzled
    int tid = threadIdx.x;
    int row0 = blockIdx.x * 32;
    stage_a(As, h1, 32, 6, row0, tid, 256);
    stage_b_sw(Bs + 0 * 4096, Wq, 64, 6, tid, 256);
    stage_b_sw(Bs + 1 * 4096, Wk, 64, 6, tid, 256);
    stage_b_sw(Bs + 2 * 4096, Wv, 64, 6, tid, 256);
    stage_b_sw(Bs + 3 * 4096, Ws, 64, 6, tid, 256);
    __syncthreads();

    int wave = tid >> 5, lane = tid & 31;
    int m = wave >> 1, rt = wave & 1;                 // 4 matrices x 2 row-tiles
    const float* bias = (m == 0) ? bq : (m == 1) ? bk : (m == 2) ? bv : bs;
    float*       out  = (m == 0) ? q  : (m == 1) ? k  : (m == 2) ? v  : hs;
    const _Float16* B  = Bs + m * 4096;
    const _Float16* Aw = As + rt * 16 * 64;

    v8f acc[4] = {};
#pragma unroll
    for (int kt = 0; kt < 2; ++kt) {
        v16h a = load_a_lds(Aw, 64, kt * 32, lane);
#pragma unroll
        for (int ct = 0; ct < 4; ++ct) {
            v16h b = load_b_frag(B, kt * 4 + ct, lane);
            acc[ct] = WMMA_F16(a, b, acc[ct]);
        }
    }
#pragma unroll
    for (int ct = 0; ct < 4; ++ct)
        store_d(out, 64, row0 + rt * 16, ct * 16, acc[ct], lane, bias);
}

// g2[N,256] = h2[N,64] @ W2[64,256]
__global__ void gemm_h2_w2(const float* __restrict__ h2, const float* __restrict__ W2,
                           float* __restrict__ g2) {
    __shared__ __align__(32) _Float16 As[32 * 64];    // 4 KB
    __shared__ __align__(32) _Float16 Bs[64 * 256];   // 32 KB swizzled
    int tid = threadIdx.x;
    int row0 = blockIdx.x * 32;
    stage_a(As, h2, 32, 6, row0, tid, 256);
    stage_b_sw(Bs, W2, 64, 8, tid, 256);
    __syncthreads();

    int wave = tid >> 5, lane = tid & 31;
    int rt = wave >> 2, ctb = (wave & 3) * 4;         // 2 row-tiles x 16 col-tiles
    const _Float16* Aw = As + rt * 16 * 64;

    v8f acc[4] = {};
#pragma unroll
    for (int kt = 0; kt < 2; ++kt) {
        v16h a = load_a_lds(Aw, 64, kt * 32, lane);
#pragma unroll
        for (int i = 0; i < 4; ++i) {
            v16h b = load_b_frag(Bs, kt * 16 + ctb + i, lane);
            acc[i] = WMMA_F16(a, b, acc[i]);
        }
    }
#pragma unroll
    for (int i = 0; i < 4; ++i)
        store_d(g2, 256, row0 + rt * 16, (ctb + i) * 16, acc[i], lane, nullptr);
}

// ---------------- utility / edge kernels -------------------------------------

__global__ void fill_f32(float* p, long long n, float val) {
    long long i = (long long)blockIdx.x * blockDim.x + threadIdx.x;
    if (i < n) p[i] = val;
}
__global__ void fill_u32(unsigned* p, long long n, unsigned val) {
    long long i = (long long)blockIdx.x * blockDim.x + threadIdx.x;
    if (i < n) p[i] = val;
}

__global__ void deg_accum(const int* __restrict__ ei, float* __restrict__ deg, int E) {
    int e = blockIdx.x * blockDim.x + threadIdx.x;
    if (e < E) atomicAdd(&deg[ei[E + e]], 1.0f);
}
__global__ void deg_to_dinv(float* p, int N) {
    int i = blockIdx.x * blockDim.x + threadIdx.x;
    if (i < N) p[i] = rsqrtf(p[i]);   // deg >= 1 always (self-loop)
}

// out_feat[dst, f] += dinv[s]*dinv[d] * src_feat[src, f]   (64 threads / edge)
__global__ void gcn_scatter64(const int* __restrict__ ei, const float* __restrict__ dinv,
                              const float* __restrict__ sf, float* __restrict__ of, int E) {
    long long t = (long long)blockIdx.x * blockDim.x + threadIdx.x;
    long long e = t >> 6; int f = (int)(t & 63);
    if (e >= E) return;
    int s = ei[e], d = ei[E + e];
    float w = dinv[s] * dinv[d];
    atomicAdd(&of[(size_t)d * 64 + f], w * sf[(size_t)s * 64 + f]);
}

// h1 = relu(h1 + dinv^2 * hx + b1)
__global__ void gcn1_finalize(float* __restrict__ h1, const float* __restrict__ hx,
                              const float* __restrict__ dinv, const float* __restrict__ b1,
                              int N) {
    long long t = (long long)blockIdx.x * blockDim.x + threadIdx.x;
    long long i = t >> 6; int f = (int)(t & 63);
    if (i >= N) return;
    float di = dinv[i];
    float r = h1[(size_t)i * 64 + f] + di * di * hx[(size_t)i * 64 + f] + b1[f];
    h1[(size_t)i * 64 + f] = r > 0.0f ? r : 0.0f;
}

// order-preserving float<->uint encoding for atomicMax running max
__device__ __forceinline__ unsigned fkey(float f) {
    unsigned u = __float_as_uint(f);
    return (u & 0x80000000u) ? ~u : (u | 0x80000000u);
}
__device__ __forceinline__ float fdec(unsigned k) {
    unsigned u = (k & 0x80000000u) ? (k & 0x7fffffffu) : ~k;
    return __uint_as_float(u);
}

// pass 1: logits[e] = dot(q[dst], k[src]) / 8 ; running max per dst  (1 wave / edge)
__global__ void attn_logits(const int* __restrict__ ei, const float* __restrict__ q,
                            const float* __restrict__ k, float* __restrict__ logits,
                            unsigned* __restrict__ mkey, int E) {
    long long wid = ((long long)blockIdx.x * blockDim.x + threadIdx.x) >> 5;
    int lane = threadIdx.x & 31;
    if (wid >= E) return;
    int s = ei[wid], d = ei[E + wid];
    const float* qp = q + (size_t)d * 64;
    const float* kp = k + (size_t)s * 64;
    float p = qp[lane] * kp[lane] + qp[lane + 32] * kp[lane + 32];
#pragma unroll
    for (int off = 16; off > 0; off >>= 1) p += __shfl_down(p, off, 32);
    if (lane == 0) {
        p *= 0.125f;   // 1/sqrt(64)
        logits[wid] = p;
        atomicMax(&mkey[d], fkey(p));
    }
}

// pass 2: e = exp(logit - m[dst]); agg[dst] += e * v[src]; denom[dst] += e
__global__ void attn_agg(const int* __restrict__ ei, const float* __restrict__ logits,
                         const unsigned* __restrict__ mkey, const float* __restrict__ v,
                         float* __restrict__ agg, float* __restrict__ denom, int E) {
    long long t = (long long)blockIdx.x * blockDim.x + threadIdx.x;
    long long e = t >> 6; int f = (int)(t & 63);
    if (e >= E) return;
    int s = ei[e], d = ei[E + e];
    float w = __expf(logits[e] - fdec(mkey[d]));
    atomicAdd(&agg[(size_t)d * 64 + f], w * v[(size_t)s * 64 + f]);
    if (f == 0) atomicAdd(&denom[d], w);
}

// h2 = agg/denom + hs  (in place in agg)
__global__ void attn_finalize(float* __restrict__ agg, const float* __restrict__ denom,
                              const float* __restrict__ hs, int N) {
    long long t = (long long)blockIdx.x * blockDim.x + threadIdx.x;
    long long i = t >> 6; int f = (int)(t & 63);
    if (i >= N) return;
    float dn = denom[i];
    dn = dn > 0.0f ? dn : 1.0f;
    agg[(size_t)i * 64 + f] = agg[(size_t)i * 64 + f] / dn + hs[(size_t)i * 64 + f];
}

// out[dst, f] += w * g2[src, f]   (1 block of 256 threads per edge)
__global__ void gcn_scatter256(const int* __restrict__ ei, const float* __restrict__ dinv,
                               const float* __restrict__ g2, float* __restrict__ out, int E) {
    int e = blockIdx.x, f = threadIdx.x;
    int s = ei[e], d = ei[E + e];
    float w = dinv[s] * dinv[d];
    atomicAdd(&out[(size_t)d * 256 + f], w * g2[(size_t)s * 256 + f]);
}

// out += dinv^2 * g2 + b2
__global__ void gcn2_finalize(float* __restrict__ out, const float* __restrict__ g2,
                              const float* __restrict__ dinv, const float* __restrict__ b2,
                              int N) {
    int i = blockIdx.x, f = threadIdx.x;
    float di = dinv[i];
    out[(size_t)i * 256 + f] += di * di * g2[(size_t)i * 256 + f] + b2[f];
}

// ---------------- launch ------------------------------------------------------

extern "C" void kernel_launch(void* const* d_in, const int* in_sizes, int n_in,
                              void* d_out, int out_size, void* d_ws, size_t ws_size,
                              hipStream_t stream) {
    const float* x  = (const float*)d_in[0];
    const int*   ei = (const int*)d_in[1];          // [2,E] row-major: src then dst
    const float* W1 = (const float*)d_in[2];  const float* b1 = (const float*)d_in[3];
    const float* Wq = (const float*)d_in[4];  const float* bq = (const float*)d_in[5];
    const float* Wk = (const float*)d_in[6];  const float* bk = (const float*)d_in[7];
    const float* Wv = (const float*)d_in[8];  const float* bv = (const float*)d_in[9];
    const float* Ws = (const float*)d_in[10]; const float* bs = (const float*)d_in[11];
    const float* W2 = (const float*)d_in[12]; const float* b2 = (const float*)d_in[13];
    float* out = (float*)d_out;

    const int N = in_sizes[0] / 256;
    const int E = in_sizes[1] / 2;
    const size_t n64 = (size_t)N * 64;

    // workspace layout (floats)
    float*    ws     = (float*)d_ws;
    float*    dinv   = ws;                       // N
    unsigned* mkey   = (unsigned*)(ws + N);      // N
    float*    denom  = ws + 2 * (size_t)N;       // N
    float*    logits = ws + 3 * (size_t)N;       // E
    float*    R      = logits + E;               // 4*N*64 contiguous
    float*    hx     = R;                        // N*64  (dead after gcn1_finalize)
    float*    h1     = R + n64;                  // N*64  (dead after gemm_qkvs)
    float*    qb     = R + 2 * n64;              // N*64  (dead after attn_logits)
    float*    kb     = R + 3 * n64;              // N*64  (dead after attn_logits)
    float*    g2     = R;                        // N*256 (reuses hx/h1/q/k)
    float*    vb     = R + 4 * n64;              // N*64
    float*    hs     = vb + n64;                 // N*64
    float*    agg    = hs + n64;                 // N*64 (becomes h2)

    const int T = 256;
    auto blk = [](long long n, int t) { return (unsigned)((n + t - 1) / t); };

    // --- degrees -> dinv ---
    fill_f32<<<blk(N, T), T, 0, stream>>>(dinv, N, 1.0f);          // self-loops
    deg_accum<<<blk(E, T), T, 0, stream>>>(ei, dinv, E);
    deg_to_dinv<<<blk(N, T), T, 0, stream>>>(dinv, N);

    // --- zero scatter targets / softmax state ---
    fill_f32<<<blk((long long)n64, T), T, 0, stream>>>(h1, (long long)n64, 0.0f);
    fill_f32<<<blk((long long)n64, T), T, 0, stream>>>(agg, (long long)n64, 0.0f);
    fill_f32<<<blk(N, T), T, 0, stream>>>(denom, N, 0.0f);
    fill_u32<<<blk(N, T), T, 0, stream>>>(mkey, N, 0u);            // key-space -inf
    fill_f32<<<blk((long long)N * 256, T), T, 0, stream>>>(out, (long long)N * 256, 0.0f);

    // --- layer 1: GCNConv 256 -> 64 ---
    gemm_x_w1<<<N / 32, 256, 0, stream>>>(x, W1, hx);
    gcn_scatter64<<<blk((long long)E * 64, T), T, 0, stream>>>(ei, dinv, hx, h1, E);
    gcn1_finalize<<<blk((long long)N * 64, T), T, 0, stream>>>(h1, hx, dinv, b1, N);

    // --- layer 2: TransformerConv 64 -> 64 ---
    gemm_qkvs<<<N / 32, 256, 0, stream>>>(h1, Wq, bq, Wk, bk, Wv, bv, Ws, bs, qb, kb, vb, hs);
    attn_logits<<<blk((long long)E * 32, T), T, 0, stream>>>(ei, qb, kb, logits, mkey, E);
    attn_agg<<<blk((long long)E * 64, T), T, 0, stream>>>(ei, logits, mkey, vb, agg, denom, E);
    attn_finalize<<<blk((long long)N * 64, T), T, 0, stream>>>(agg, denom, hs, N);

    // --- layer 3: GCNConv 64 -> 256 ---
    gemm_h2_w2<<<N / 32, 256, 0, stream>>>(agg, W2, g2);
    gcn_scatter256<<<(unsigned)E, 256, 0, stream>>>(ei, dinv, g2, out, E);
    gcn2_finalize<<<(unsigned)N, 256, 0, stream>>>(out, g2, dinv, b2, N);
}
